// GATLayer_12386685682072
// MI455X (gfx1250) — compile-verified, matrix operand order checked by reading, not run on hardware
//
#include <hip/hip_runtime.h>
#include <hip/hip_bf16.h>

typedef __attribute__((ext_vector_type(16))) _Float16 v16h;
typedef __attribute__((ext_vector_type(8)))  float    v8f;

#define N_NODES 4096
#define H_NUM   8
#define LOG2E   1.4426950408889634f

// ---------------------------------------------------------------------------
// Phase 1: feats = X@W (4096x64 @ 64x64), attention logits (pre-scaled by
// log2(e) so phase 2 can use native v_exp_f32 = exp2 directly), and pack
// feats into f16 WMMA B-matrix register layout per head:
//   wsB[h][jt][lane][v][slot], jt = j/32, k = j%32,
//   half = (k>>3)&1, lane = d + 16*half,
//   v = ((k>>4)<<2) + ((k&7)>>1), slot = k&1
// Column 8 is set to 1.0 (not 0): the WMMA then produces the softmax
// denominator in D[:,8] for free. Columns 9..15 are zero.
// ---------------------------------------------------------------------------
__global__ __launch_bounds__(64) void gat_phase1(
    const float* __restrict__ X, const float* __restrict__ W,
    const float* __restrict__ att_self, const float* __restrict__ att_neigh,
    float* __restrict__ aself_t, float* __restrict__ aneigh_t,
    _Float16* __restrict__ wsB)
{
    __shared__ float xrow[64];
    __shared__ float ps[64];
    __shared__ float pn[64];
    const int n = blockIdx.x;
    const int d = threadIdx.x;

    xrow[d] = X[n * 64 + d];
    __syncthreads();

    float acc = 0.f;
    #pragma unroll
    for (int f = 0; f < 64; ++f)
        acc = fmaf(xrow[f], W[f * 64 + d], acc);

    const int h = d >> 3, dd = d & 7;
    ps[d] = acc * att_self[h * 8 + dd];
    pn[d] = acc * att_neigh[h * 8 + dd];
    __syncthreads();

    if (d < 8) {
        float s = 0.f, t = 0.f;
        #pragma unroll
        for (int q = 0; q < 8; ++q) { s += ps[d * 8 + q]; t += pn[d * 8 + q]; }
        aself_t[d * N_NODES + n]  = s * LOG2E;   // head-major, log2e-prescaled
        aneigh_t[d * N_NODES + n] = t * LOG2E;
    }

    // pack this feat element (node n = k-index, head h, dim dd = column)
    const int jt   = n >> 5;
    const int k    = n & 31;
    const int half = (k >> 3) & 1;
    const int v    = ((k >> 4) << 2) + ((k & 7) >> 1);
    const int slot = k & 1;
    const int lane = dd + (half << 4);
    const size_t base  = ((((size_t)h * 128 + jt) * 32 + lane)     * 8 + v) * 2 + slot;
    const size_t basez = ((((size_t)h * 128 + jt) * 32 + (lane+8)) * 8 + v) * 2 + slot;
    wsB[base]  = (_Float16)acc;
    // column dd+8: ones-column at n==8 gives denominator via WMMA, rest zero
    wsB[basez] = (dd == 0) ? (_Float16)1.f : (_Float16)0.f;
}

// ---------------------------------------------------------------------------
// Phase 2: fused masked-softmax + aggregation.
// Block = 16-row tile of nodes, 8 waves = 8 heads. A tile staged in LDS once
// per chunk, shared by all heads. One-pass softmax: the WMMA accumulates both
// the unnormalized numerator (cols 0..7) and the denominator (col 8, via the
// ones-column in B); divide once at the end. B fragments for the whole chunk
// are loaded up-front so the WMMAs never stall on a fresh global load.
// ---------------------------------------------------------------------------
__global__ __launch_bounds__(256) void gat_phase2(
    const float* __restrict__ A,
    const float* __restrict__ aself_t, const float* __restrict__ aneigh_t,
    const _Float16* __restrict__ wsB, const float* __restrict__ bias,
    float* __restrict__ out)
{
    __shared__ __align__(16) float ldsA[16 * 128];   // 8 KB A tile
    __shared__ float ldsDen[H_NUM * 16];

    const int it   = blockIdx.x;
    const int tid  = threadIdx.x;
    const int h    = tid >> 5;        // wave id = head
    const int lane = tid & 31;
    const int m    = lane & 15;       // row within tile
    const int kb0  = (lane < 16) ? 0 : 8;  // WMMA A-layout K base for this half

    const float  s_i  = aself_t[h * N_NODES + it * 16 + m];  // log2e-prescaled
    const float* tptr = aneigh_t + h * N_NODES;
    const v16h*  Bv   = (const v16h*)wsB;

    v8f c = {0.f, 0.f, 0.f, 0.f, 0.f, 0.f, 0.f, 0.f};

    const float4* A4 = (const float4*)A;

    for (int jc = 0; jc < N_NODES; jc += 128) {
        // Issue all four B-fragment loads for this chunk up-front: their
        // latency is hidden behind LDS staging + barrier + first exp chain.
        const size_t bbase = ((size_t)h * 128 + (jc >> 5)) * 32 + lane;
        v16h bf[4];
        bf[0] = Bv[bbase];
        bf[1] = Bv[bbase + 32];
        bf[2] = Bv[bbase + 64];
        bf[3] = Bv[bbase + 96];

        // cooperative stage: 16 rows x 128 cols f32 = 512 float4s, 256 threads
        float4* lds4 = (float4*)ldsA;
        #pragma unroll
        for (int q = tid; q < 512; q += 256) {
            const int row = q >> 5, c4 = q & 31;
            lds4[q] = A4[(it * 16 + row) * 1024 + (jc >> 2) + c4];
        }
        if (jc + 128 < N_NODES) {
            const int row = (tid >> 4) & 15, c4 = ((tid & 15) << 1);
            __builtin_prefetch(&A4[(it * 16 + row) * 1024 + ((jc + 128) >> 2) + c4], 0, 1);
        }
        __syncthreads();

        #pragma unroll
        for (int sub = 0; sub < 4; ++sub) {
            const int col0 = sub * 32 + kb0;       // K run 0: col0..col0+7
            const int col1 = col0 + 16;            // K run 1: col1..col1+7

            float av[16], tv[16];
            {
                float4 x0 = *(const float4*)&ldsA[m * 128 + col0];
                float4 x1 = *(const float4*)&ldsA[m * 128 + col0 + 4];
                float4 x2 = *(const float4*)&ldsA[m * 128 + col1];
                float4 x3 = *(const float4*)&ldsA[m * 128 + col1 + 4];
                av[0]=x0.x; av[1]=x0.y; av[2]=x0.z; av[3]=x0.w;
                av[4]=x1.x; av[5]=x1.y; av[6]=x1.z; av[7]=x1.w;
                av[8]=x2.x; av[9]=x2.y; av[10]=x2.z; av[11]=x2.w;
                av[12]=x3.x; av[13]=x3.y; av[14]=x3.z; av[15]=x3.w;
                float4 y0 = *(const float4*)&tptr[jc + col0];
                float4 y1 = *(const float4*)&tptr[jc + col0 + 4];
                float4 y2 = *(const float4*)&tptr[jc + col1];
                float4 y3 = *(const float4*)&tptr[jc + col1 + 4];
                tv[0]=y0.x; tv[1]=y0.y; tv[2]=y0.z; tv[3]=y0.w;
                tv[4]=y1.x; tv[5]=y1.y; tv[6]=y1.z; tv[7]=y1.w;
                tv[8]=y2.x; tv[9]=y2.y; tv[10]=y2.z; tv[11]=y2.w;
                tv[12]=y3.x; tv[13]=y3.y; tv[14]=y3.z; tv[15]=y3.w;
            }

            v16h afrag;
            #pragma unroll
            for (int i = 0; i < 16; ++i) {
                // x is already in log2 domain; lrelu commutes with the
                // positive scale, so exp(lrelu(s+t)) == exp2(lrelu(x))
                float x = s_i + tv[i];
                x = fmaxf(x, 0.f) + 0.2f * fminf(x, 0.f);   // leaky_relu(0.2)
                float e = av[i] * __builtin_amdgcn_exp2f(x); // A in {0,1}: mask
                afrag[i] = (_Float16)e;
            }

            c = __builtin_amdgcn_wmma_f32_16x16x32_f16(
                    false, afrag, false, bf[sub], (short)0, c, false, false);
        }
        __syncthreads();
    }

    // C/D layout: VGPR r -> M = r + 8*(lane>=16), N = lane&15.
    // Column 8 (lanes 8 and 24) holds the row denominators: spill to LDS.
    const int ncol = lane & 15;
    if (ncol == 8) {
        #pragma unroll
        for (int r = 0; r < 8; ++r)
            ldsDen[h * 16 + r + ((lane >> 4) << 3)] = c[r];
    }
    __syncthreads();

    if (ncol < 8) {
        const float b = bias[h * 8 + ncol];
        #pragma unroll
        for (int r = 0; r < 8; ++r) {
            const int M = r + ((lane >> 4) << 3);
            const float val = c[r] / ldsDen[h * 16 + M] + b;
            out[(it * 16 + M) * 64 + h * 8 + ncol] = fmaxf(val, 0.f);
        }
    }
}

// ---------------------------------------------------------------------------
extern "C" void kernel_launch(void* const* d_in, const int* in_sizes, int n_in,
                              void* d_out, int out_size, void* d_ws, size_t ws_size,
                              hipStream_t stream) {
    const float* X    = (const float*)d_in[0];
    const float* A    = (const float*)d_in[1];
    const float* W    = (const float*)d_in[2];
    const float* as   = (const float*)d_in[3];
    const float* an   = (const float*)d_in[4];
    const float* bias = (const float*)d_in[5];
    float* out = (float*)d_out;

    // workspace layout: [0,1MB) packed f16 B-matrices; then a_self_t, a_neigh_t
    _Float16* wsB      = (_Float16*)d_ws;
    float*    aself_t  = (float*)((char*)d_ws + (1u << 20));
    float*    aneigh_t = aself_t + H_NUM * N_NODES;

    gat_phase1<<<N_NODES, 64, 0, stream>>>(X, W, as, an, aself_t, aneigh_t, wsB);
    gat_phase2<<<N_NODES / 16, 256, 0, stream>>>(A, aself_t, aneigh_t, wsB, bias, out);
}
